// MultiHeadAttn_23390391894761
// MI455X (gfx1250) — compile-verified, matrix operand order checked by reading, not run on hardware
//
#include <hip/hip_runtime.h>

// ---------------------------------------------------------------------------
// MultiHeadAttn for MI455X (gfx1250): bf16 WMMA projections + flash attention.
// B=2, T=2048, D=1024, H=16, dh=64. All matmuls via v_wmma_f32_16x16x32_bf16.
// Attention K/V tiles staged with GLOBAL_LOAD_ASYNC_TO_LDS_B128 (ASYNCcnt).
// Workspace: q,k bf16 [B,H,T,64]; v bf16 [B,H,64,T] (pre-transposed). 24 MB.
// ---------------------------------------------------------------------------

typedef __attribute__((ext_vector_type(16))) __bf16 v16bf;
typedef __attribute__((ext_vector_type(8)))  __bf16 v8bf;
typedef __attribute__((ext_vector_type(2)))  __bf16 v2bf;
typedef __attribute__((ext_vector_type(8)))  float  v8f;

#define BATCH 2
#define TSEQ  2048
#define DDIM  1024
#define NHEAD 16
#define DH    64

// float -> bf16 via hardware convert (v_cvt_pk_bf16_f32 on gfx1250)
__device__ __forceinline__ unsigned pack2bf(float x, float y) {
#if __has_builtin(__builtin_amdgcn_cvt_pk_bf16_f32)
  return __builtin_bit_cast(unsigned, __builtin_amdgcn_cvt_pk_bf16_f32(x, y));
#else
  v2bf p = {(__bf16)x, (__bf16)y};
  return __builtin_bit_cast(unsigned, p);
#endif
}

__device__ __forceinline__ unsigned short bf_bits(float f) {
  __bf16 h = (__bf16)f;
  return __builtin_bit_cast(unsigned short, h);
}

__device__ __forceinline__ v16bf ld_frag(const unsigned short* p_lo,
                                         const unsigned short* p_hi) {
  v8bf lo = *reinterpret_cast<const v8bf*>(p_lo);   // ds_load_b128
  v8bf hi = *reinterpret_cast<const v8bf*>(p_hi);   // ds_load_b128
  return __builtin_shufflevector(lo, hi, 0,1,2,3,4,5,6,7,8,9,10,11,12,13,14,15);
}

__device__ __forceinline__ v8f wmma_bf16(v16bf a, v16bf b, v8f c) {
  return __builtin_amdgcn_wmma_f32_16x16x32_bf16(false, a, false, b,
                                                 (short)0, c, false, false);
}

// async copy 16B global -> LDS (GV mode), tracked by ASYNCcnt
__device__ __forceinline__ void async_g2l_b128(void* lds_dst, const void* gsrc) {
  unsigned dst = (unsigned)(unsigned long long)lds_dst;   // LDS offset = addr[31:0]
  unsigned long long src = (unsigned long long)gsrc;
  asm volatile("global_load_async_to_lds_b128 %0, %1, off"
               :: "v"(dst), "v"(src) : "memory");
}
__device__ __forceinline__ void wait_async0() {
  asm volatile("s_wait_asynccnt 0x0" ::: "memory");
}

// ---------------------------------------------------------------------------
// Kernel 1: fused QKV projection. [4096,1024]x[1024,1024] GEMM in bf16 WMMA,
// fp32 inputs converted in flight. z selects Q/K/V. Q,K out: bf16 [B,H,T,64];
// V out: bf16 [B,H,64,T] (transposed for direct LDS tile loads in attention).
// ---------------------------------------------------------------------------
__global__ __launch_bounds__(256) void proj_qkv_kernel(
    const float* __restrict__ Xq, const float* __restrict__ Xk,
    const float* __restrict__ Xv,
    const float* __restrict__ Wq, const float* __restrict__ Wk,
    const float* __restrict__ Wv,
    unsigned short* __restrict__ Oq, unsigned short* __restrict__ Ok,
    unsigned short* __restrict__ Ov) {
  __shared__ unsigned short As[128][40];   // [row][k] bf16, pad keeps 16B align
  __shared__ unsigned short Bs[128][40];   // [col][k] bf16 (W tile transposed)

  const int z = blockIdx.z;
  const float* __restrict__ X = (z == 0) ? Xq : (z == 1) ? Xk : Xv;
  const float* __restrict__ W = (z == 0) ? Wq : (z == 1) ? Wk : Wv;
  unsigned short* __restrict__ O = (z == 0) ? Oq : (z == 1) ? Ok : Ov;

  const int tid  = threadIdx.x;
  const int lane = tid & 31;
  const int wid  = tid >> 5;     // 8 waves
  const int mw   = wid >> 2;     // 0..1 : 64-row strip
  const int nw   = wid & 3;      // 0..3 : 32-col strip
  const int hh   = lane >> 4;    // half-wave (K-chunk select per ISA layout)
  const int ln16 = lane & 15;

  const int rowBase = blockIdx.y * 128;
  const int colBase = blockIdx.x * 128;

  const v8f z8 = {0.f, 0.f, 0.f, 0.f, 0.f, 0.f, 0.f, 0.f};
  v8f acc[4][2];
#pragma unroll
  for (int mt = 0; mt < 4; ++mt)
#pragma unroll
    for (int nt = 0; nt < 2; ++nt) acc[mt][nt] = z8;

  for (int k0 = 0; k0 < DDIM; k0 += 32) {
    // ---- issue ALL tile loads first (8x global_load_b128 in flight) ----
    float4 fa[4], fb[4];
#pragma unroll
    for (int i = 0; i < 4; ++i) {
      int idx = tid + i * 256;          // 0..1023
      int r   = idx >> 3;               // 0..127
      int c4  = idx & 7;                // 0..7 float4s
      fa[i] = *reinterpret_cast<const float4*>(
          &X[(size_t)(rowBase + r) * DDIM + k0 + c4 * 4]);
    }
#pragma unroll
    for (int i = 0; i < 4; ++i) {
      int idx = tid + i * 256;
      int kr  = idx >> 5;               // 0..31
      int c4  = idx & 31;               // 0..31 float4s
      fb[i] = *reinterpret_cast<const float4*>(
          &W[(size_t)(k0 + kr) * DDIM + colBase + c4 * 4]);
    }
    if (k0 + 32 < DDIM) {  // gfx1250 global_prefetch_b8 for next K slab
      __builtin_prefetch(&X[(size_t)(rowBase + (tid >> 1)) * DDIM + k0 + 32], 0, 3);
      __builtin_prefetch(&W[(size_t)(k0 + 32 + (tid >> 3)) * DDIM + colBase], 0, 3);
    }
    // ---- convert + store to LDS ----
#pragma unroll
    for (int i = 0; i < 4; ++i) {
      int idx = tid + i * 256;
      int r   = idx >> 3;
      int c4  = idx & 7;
      unsigned long long p =  (unsigned long long)pack2bf(fa[i].x, fa[i].y)
                           | ((unsigned long long)pack2bf(fa[i].z, fa[i].w) << 32);
      *reinterpret_cast<unsigned long long*>(&As[r][c4 * 4]) = p;
    }
#pragma unroll
    for (int i = 0; i < 4; ++i) {
      int idx = tid + i * 256;
      int kr  = idx >> 5;
      int c4  = idx & 31;
      Bs[c4 * 4 + 0][kr] = bf_bits(fb[i].x);
      Bs[c4 * 4 + 1][kr] = bf_bits(fb[i].y);
      Bs[c4 * 4 + 2][kr] = bf_bits(fb[i].z);
      Bs[c4 * 4 + 3][kr] = bf_bits(fb[i].w);
    }
    __syncthreads();

    v16bf a[4], b[2];
#pragma unroll
    for (int mt = 0; mt < 4; ++mt) {
      int r = mw * 64 + mt * 16 + ln16;
      a[mt] = ld_frag(&As[r][8 * hh], &As[r][16 + 8 * hh]);
    }
#pragma unroll
    for (int nt = 0; nt < 2; ++nt) {
      int n = nw * 32 + nt * 16 + ln16;
      b[nt] = ld_frag(&Bs[n][8 * hh], &Bs[n][16 + 8 * hh]);
    }
#pragma unroll
    for (int mt = 0; mt < 4; ++mt)
#pragma unroll
      for (int nt = 0; nt < 2; ++nt)
        acc[mt][nt] = wmma_bf16(a[mt], b[nt], acc[mt][nt]);
    __syncthreads();
  }

  // epilogue: C layout M = vgpr + 8*(lane>=16), N = lane%16.
  // Q,K -> [B,H,T,64]; V -> [B,H,64,T] (transposed).
#pragma unroll
  for (int mt = 0; mt < 4; ++mt)
#pragma unroll
    for (int nt = 0; nt < 2; ++nt)
#pragma unroll
      for (int r = 0; r < 8; ++r) {
        int m  = rowBase + mw * 64 + mt * 16 + r + 8 * hh;
        int n  = colBase + nw * 32 + nt * 16 + ln16;
        int bb = m >> 11, t = m & (TSEQ - 1);
        int hd = n >> 6,  d = n & (DH - 1);
        size_t oidx;
        if (z == 2)
          oidx = (((size_t)(bb * NHEAD + hd) * DH) + d) * TSEQ + t;
        else
          oidx = (((size_t)(bb * NHEAD + hd) * TSEQ) + t) * DH + d;
        O[oidx] = bf_bits(acc[mt][nt][r]);
      }
}

// ---------------------------------------------------------------------------
// Kernel 2: flash attention. One block per (b, h, 64 q-rows); 4 waves x 16
// rows. 64-key blocks, online softmax in f32, bf16 WMMA for QK^T and P*V.
// K/V tiles staged via async global->LDS DMA (ASYNCcnt).
// ---------------------------------------------------------------------------
__global__ __launch_bounds__(128) void fattn_kernel(
    const unsigned short* __restrict__ Qb, const unsigned short* __restrict__ Kb,
    const unsigned short* __restrict__ Vb,
    const int* __restrict__ Qlen, const int* __restrict__ Vlen,
    float* __restrict__ Out) {
  __shared__ unsigned short Qs[64][72];        // [q][d]
  __shared__ unsigned short Ks[64][72];        // [key][d]
  __shared__ unsigned short Vt[64][72];        // [d][key]
  __shared__ unsigned short Ps[4][16][72];     // per-wave P scratch [q][key]

  const int tid  = threadIdx.x;
  const int lane = tid & 31;
  const int wid  = tid >> 5;    // 0..3
  const int hh   = lane >> 4;
  const int ln16 = lane & 15;

  const int h  = blockIdx.y;
  const int bz = blockIdx.z;
  const int qlen = Qlen[bz];
  const int vlen = Vlen[bz];

  const size_t headBase = (size_t)(bz * NHEAD + h) * TSEQ * DH;  // q,k layout
  const int qbase = blockIdx.x * 64;

  // stage Q tile once (async DMA, no VGPR round-trip)
#pragma unroll
  for (int i = 0; i < 4; ++i) {
    int idx = tid + i * 128;
    int r = idx >> 3, c = idx & 7;
    async_g2l_b128(&Qs[r][c * 8],
                   &Qb[headBase + (size_t)(qbase + r) * DH + c * 8]);
  }
  wait_async0();
  __syncthreads();

  v16bf qf[2];
#pragma unroll
  for (int dc = 0; dc < 2; ++dc) {
    int r = wid * 16 + ln16;
    qf[dc] = ld_frag(&Qs[r][dc * 32 + 8 * hh], &Qs[r][dc * 32 + 16 + 8 * hh]);
  }

  const v8f z8 = {0.f, 0.f, 0.f, 0.f, 0.f, 0.f, 0.f, 0.f};
  float mrow[8], lrow[8];
  v8f o[4];
#pragma unroll
  for (int r = 0; r < 8; ++r) { mrow[r] = -3.0e38f; lrow[r] = 0.f; }
#pragma unroll
  for (int dt = 0; dt < 4; ++dt) o[dt] = z8;

  for (int kb0 = 0; kb0 < TSEQ; kb0 += 64) {
    __syncthreads();
    // stage K tile [key][d] and V tile [d][key] via async global->LDS copy
#pragma unroll
    for (int i = 0; i < 4; ++i) {
      int idx = tid + i * 128;
      int r = idx >> 3, c = idx & 7;
      async_g2l_b128(&Ks[r][c * 8],
                     &Kb[headBase + (size_t)(kb0 + r) * DH + c * 8]);
      async_g2l_b128(&Vt[r][c * 8],
                     &Vb[headBase + (size_t)r * TSEQ + kb0 + c * 8]);
    }
    wait_async0();
    __syncthreads();

    // ---- S = (Q K^T) * 1/sqrt(dh), with key-length mask ----
    float p[4][8];
    float tmax[8];
#pragma unroll
    for (int r = 0; r < 8; ++r) tmax[r] = -3.0e38f;
#pragma unroll
    for (int kt = 0; kt < 4; ++kt) {
      v8f s = z8;
#pragma unroll
      for (int dc = 0; dc < 2; ++dc) {
        int n = kt * 16 + ln16;
        v16bf kfrag = ld_frag(&Ks[n][dc * 32 + 8 * hh],
                              &Ks[n][dc * 32 + 16 + 8 * hh]);
        s = wmma_bf16(qf[dc], kfrag, s);
      }
      int col = kb0 + kt * 16 + ln16;
      bool masked = (col >= vlen);
#pragma unroll
      for (int r = 0; r < 8; ++r) {
        float sv = s[r] * 0.125f;          // 1/sqrt(64)
        if (masked) sv = -1.0e12f;         // additive key mask
        p[kt][r] = sv;
        tmax[r] = fmaxf(tmax[r], sv);
      }
    }
    // row max across 16-lane column groups
#pragma unroll
    for (int r = 0; r < 8; ++r) {
      float v = tmax[r];
      v = fmaxf(v, __shfl_xor(v, 1));
      v = fmaxf(v, __shfl_xor(v, 2));
      v = fmaxf(v, __shfl_xor(v, 4));
      v = fmaxf(v, __shfl_xor(v, 8));
      tmax[r] = v;
    }
    float scale[8];
#pragma unroll
    for (int r = 0; r < 8; ++r) {
      float mn = fmaxf(mrow[r], tmax[r]);
      scale[r] = __expf(mrow[r] - mn);
      mrow[r] = mn;
    }
    float rsum[8];
#pragma unroll
    for (int r = 0; r < 8; ++r) rsum[r] = 0.f;
#pragma unroll
    for (int kt = 0; kt < 4; ++kt)
#pragma unroll
      for (int r = 0; r < 8; ++r) {
        float e = __expf(p[kt][r] - mrow[r]);
        p[kt][r] = e;
        rsum[r] += e;
      }
#pragma unroll
    for (int r = 0; r < 8; ++r) {
      float v = rsum[r];
      v += __shfl_xor(v, 1);
      v += __shfl_xor(v, 2);
      v += __shfl_xor(v, 4);
      v += __shfl_xor(v, 8);
      lrow[r] = lrow[r] * scale[r] + v;
    }
#pragma unroll
    for (int dt = 0; dt < 4; ++dt)
#pragma unroll
      for (int r = 0; r < 8; ++r) o[dt][r] *= scale[r];

    // re-layout P (C-layout regs -> A-layout frags) through per-wave LDS;
    // DS ops are in-order within a wave, no barrier needed.
#pragma unroll
    for (int kt = 0; kt < 4; ++kt)
#pragma unroll
      for (int r = 0; r < 8; ++r)
        Ps[wid][r + 8 * hh][kt * 16 + ln16] = bf_bits(p[kt][r]);

    v16bf pf[2];
#pragma unroll
    for (int kc = 0; kc < 2; ++kc)
      pf[kc] = ld_frag(&Ps[wid][ln16][kc * 32 + 8 * hh],
                       &Ps[wid][ln16][kc * 32 + 16 + 8 * hh]);

    // ---- O += P V ----
#pragma unroll
    for (int dt = 0; dt < 4; ++dt)
#pragma unroll
      for (int kc = 0; kc < 2; ++kc) {
        int d = dt * 16 + ln16;
        v16bf vfrag = ld_frag(&Vt[d][kc * 32 + 8 * hh],
                              &Vt[d][kc * 32 + 16 + 8 * hh]);
        o[dt] = wmma_bf16(pf[kc], vfrag, o[dt]);
      }
  }

  // epilogue: normalize by row-sum, apply query-length mask, fp32 out
#pragma unroll
  for (int r = 0; r < 8; ++r) {
    int t = qbase + wid * 16 + r + 8 * hh;
    float inv = 1.0f / lrow[r];
    float qm = (t < qlen) ? 1.0f : 0.0f;
#pragma unroll
    for (int dt = 0; dt < 4; ++dt) {
      int d = dt * 16 + ln16;
      Out[((size_t)(bz * TSEQ + t)) * (NHEAD * DH) + h * DH + d] =
          o[dt][r] * inv * qm;
    }
  }
}

// ---------------------------------------------------------------------------
extern "C" void kernel_launch(void* const* d_in, const int* in_sizes, int n_in,
                              void* d_out, int out_size, void* d_ws,
                              size_t ws_size, hipStream_t stream) {
  (void)in_sizes; (void)n_in; (void)out_size; (void)ws_size;
  const float* Qseq = (const float*)d_in[0];
  const float* Kseq = (const float*)d_in[1];
  const float* Vseq = (const float*)d_in[2];
  const int*   Ql   = (const int*)d_in[3];
  const int*   Vl   = (const int*)d_in[4];
  const float* WQ   = (const float*)d_in[5];
  const float* WK   = (const float*)d_in[6];
  const float* WV   = (const float*)d_in[7];
  float* Out = (float*)d_out;

  // workspace: 3 bf16 buffers of B*H*T*64 elements (8 MB each, 24 MB total)
  unsigned short* qb = (unsigned short*)d_ws;
  const size_t per = (size_t)BATCH * NHEAD * TSEQ * DH;
  unsigned short* kb = qb + per;
  unsigned short* vb = kb + per;

  dim3 gA(DDIM / 128, (BATCH * TSEQ) / 128, 3);   // (8, 32, 3)
  proj_qkv_kernel<<<gA, 256, 0, stream>>>(Qseq, Kseq, Vseq, WQ, WK, WV,
                                          qb, kb, vb);
  dim3 gB(TSEQ / 64, NHEAD, BATCH);               // (32, 16, 2)
  fattn_kernel<<<gB, 128, 0, stream>>>(qb, kb, vb, Ql, Vl, Out);
}
